// Net_89361089560891
// MI455X (gfx1250) — compile-verified, matrix operand order checked by reading, not run on hardware
//
#include <hip/hip_runtime.h>

// ------------------------------------------------------------------
// Problem constants (from reference): B=256, N=128, S=4, F_NODE=11,
// F_HID=128, N_LAYERS=4, N_OUT=19.  K padded to 128 for all layers.
// ------------------------------------------------------------------
#define NB    256
#define NN    128
#define FH    128
#define FNODE 11
#define NOUT  19
#define BSTRIDE 40   // padded f16 row stride for LDS B-stage (bank-conflict-free)

typedef _Float16 h8   __attribute__((ext_vector_type(8)));
typedef _Float16 v16h __attribute__((ext_vector_type(16)));
typedef float    v8f  __attribute__((ext_vector_type(8)));
typedef int      vsi4 __attribute__((vector_size(16)));   // int4 for async-LDS builtin

// Load a 16x32 f16 WMMA A-fragment from a row-major matrix (ld in elems).
// Per CDNA5 ISA: lanes 0-15 hold row=l, K={0..7,16..23}; lanes 16-31 hold
// row=l, K={8..15,24..31}.  B-fragments use the same pattern on a transposed
// (column-contiguous) operand.
__device__ __forceinline__ v16h frag_rm(const _Float16* __restrict__ base,
                                        int ld, int row0, int k0, int lane) {
  const int l  = lane & 15;
  const int hi = lane >> 4;
  const _Float16* p = base + (size_t)(row0 + l) * ld + k0 + hi * 8;
  h8 x0 = *(const h8*)(p);
  h8 x1 = *(const h8*)(p + 16);
  v16h r;
#pragma unroll
  for (int i = 0; i < 8; ++i) { r[i] = x0[i]; r[8 + i] = x1[i]; }
  return r;
}

// B-fragment gather from the LDS stage (rows = output col, padded stride).
__device__ __forceinline__ v16h frag_lds(const _Float16* Bs, int row0, int lane) {
  const int l  = lane & 15;
  const int hi = lane >> 4;
  const _Float16* p = Bs + (row0 + l) * BSTRIDE + hi * 8;
  h8 x0 = *(const h8*)(p);
  h8 x1 = *(const h8*)(p + 16);
  v16h r;
#pragma unroll
  for (int i = 0; i < 8; ++i) { r[i] = x0[i]; r[8 + i] = x1[i]; }
  return r;
}

// Async global->LDS copy of 16B, falling back to a VGPR round-trip when the
// gfx1250 async builtin is unavailable.  LDS addrspace(3) offset is the low
// 32 bits of the generic pointer (AMDGPU generic->local lowering).
__device__ __forceinline__ void stage16(const _Float16* src, _Float16* dst) {
#if defined(__gfx1250__) && \
    __has_builtin(__builtin_amdgcn_global_load_async_to_lds_b128)
  typedef __attribute__((address_space(1))) vsi4 g_vsi4;
  typedef __attribute__((address_space(3))) vsi4 l_vsi4;
  g_vsi4* g = (g_vsi4*)(unsigned long long)(const void*)src;
  l_vsi4* l = (l_vsi4*)(unsigned int)(unsigned long long)(void*)dst;
  __builtin_amdgcn_global_load_async_to_lds_b128(g, l, 0, 0);
#else
  *(h8*)dst = *(const h8*)src;
#endif
}

__device__ __forceinline__ void stage_fence() {
#if defined(__gfx1250__) && \
    __has_builtin(__builtin_amdgcn_global_load_async_to_lds_b128)
#if __has_builtin(__builtin_amdgcn_s_wait_asynccnt)
  __builtin_amdgcn_s_wait_asynccnt(0);
#else
  asm volatile("s_wait_asynccnt 0" ::: "memory");
#endif
#endif
}

// ------------------------------------------------------------------
// GEMM1: TU(32768 x 640) = X(32768 x 128) @ W.  WT stored (640 x 128)
// transposed so B-fragments load contiguously.  A fragments hoisted into
// registers (loaded once, reused for all 5 column tiles).  Epilogue
// scatters into transposed aggregator operands TT[b][o][j*4+s], UT[b][o][j].
// Grid: 2048 blocks x 256 thr (8 waves); wave does 5 col-tiles, 4 wmma each.
// ------------------------------------------------------------------
__global__ void __launch_bounds__(256)
gemm1_transform(const _Float16* __restrict__ X,
                const _Float16* __restrict__ WT,
                _Float16* __restrict__ TT,
                _Float16* __restrict__ UT) {
  const int lane  = threadIdx.x & 31;
  const int wave  = threadIdx.x >> 5;
  const int row0  = blockIdx.x * 16;
  const int batch = row0 >> 7;
  const int j0    = row0 & 127;

  v16h afr[4];
#pragma unroll
  for (int kk = 0; kk < 4; ++kk)
    afr[kk] = frag_rm(X, 128, row0, kk * 32, lane);

  for (int t = 0; t < 5; ++t) {
    const int c0 = (wave + t * 8) * 16;   // 0..624
    v8f acc = {};
#pragma unroll
    for (int kk = 0; kk < 4; ++kk) {
      v16h b = frag_rm(WT, 128, c0, kk * 32, lane);
      acc = __builtin_amdgcn_wmma_f32_16x16x32_f16(false, afr[kk], false, b,
                                                   (short)0, acc, false, false);
    }
    const int l  = lane & 15;
    const int hi = lane >> 4;
    const int jb = j0 + hi * 8;
    if (c0 < 512) {                       // T channels: dst[o][j*4+s]
      const int s = c0 >> 7;
      const int o = (c0 & 127) + l;
      _Float16* dst = TT + (size_t)batch * (128 * 512) + (size_t)o * 512 + s;
#pragma unroll
      for (int v = 0; v < 8; ++v) dst[(size_t)(jb + v) * 4] = (_Float16)acc[v];
    } else {                              // U channel: dst[o][j]
      const int o = (c0 - 512) + l;
      _Float16* dst = UT + (size_t)batch * (128 * 128) + (size_t)o * 128;
#pragma unroll
      for (int v = 0; v < 8; ++v) dst[jb + v] = (_Float16)acc[v];
    }
  }
}

// ------------------------------------------------------------------
// GEMM2 helper: one accumulation term.  K-outer / column-inner: the A
// fragment is loaded once per K-step and reused by all 8 column tiles;
// the 128x32 B chunk is staged in LDS once per block (shared by 8 waves)
// via async global->LDS loads, then gathered with ds_load_b128.
// ------------------------------------------------------------------
__device__ __forceinline__ void gemm2_term(const _Float16* __restrict__ Asrc,
                                           const _Float16* __restrict__ Bsrc,
                                           int ld, int nsteps,
                                           _Float16* Bs, v8f (&acc)[8],
                                           int tid) {
  const int lane  = tid & 31;
  const int wave  = tid >> 5;
  const int i0    = wave * 16;
  const int srow  = tid >> 1;          // 0..127: B row staged by this thread
  const int shalf = (tid & 1) * 16;    // 0 or 16: which 16-elem half

  for (int kk = 0; kk < nsteps; ++kk) {
    const int k0 = kk * 32;
    // Stage B chunk: rows 0..127 (output col), cols k0..k0+31.
    const _Float16* src = Bsrc + (size_t)srow * ld + k0 + shalf;
    _Float16* dst = &Bs[srow * BSTRIDE + shalf];
    stage16(src, dst);
    stage16(src + 8, dst + 8);
    stage_fence();                      // drain own ASYNCcnt before barrier
    __syncthreads();
    v16h a = frag_rm(Asrc, ld, i0, k0, lane);
#pragma unroll
    for (int c = 0; c < 8; ++c) {
      v16h bb = frag_lds(Bs, c * 16, lane);
      acc[c] = __builtin_amdgcn_wmma_f32_16x16x32_f16(false, a, false, bb,
                                                      (short)0, acc[c],
                                                      false, false);
    }
    __syncthreads();
  }
}

// ------------------------------------------------------------------
// GEMM2 (per batch): Out_b = relu(E_b@T_b + A_b@U_b + X_b@root + bias),
// written as f16 for the next layer.  Grid: 256 blocks x 256 thr; wave w
// owns row-tile w and keeps 8 accumulator tiles (64 VGPRs) live across
// all 24 K-steps (16 edge + 4 kbias + 4 root).
// ------------------------------------------------------------------
__global__ void __launch_bounds__(256)
gemm2_aggregate(const _Float16* __restrict__ E,    // (B,128,512) row-major
                const _Float16* __restrict__ A,    // (B,128,128)
                const _Float16* __restrict__ TT,   // (B,128,512) [o][j*4+s]
                const _Float16* __restrict__ UT,   // (B,128,128) [o][j]
                const _Float16* __restrict__ Xin,  // (B,128,128)
                const _Float16* __restrict__ RwT,  // (128,128)   [o][f]
                const float* __restrict__ bias,    // (128)
                _Float16* __restrict__ Xout) {     // (B,128,128)
  __shared__ _Float16 Bs[128 * BSTRIDE];           // 10 KB stage
  const int b    = blockIdx.x;
  const int tid  = threadIdx.x;
  const int lane = tid & 31;
  const int wave = tid >> 5;
  const int i0   = wave * 16;
  const _Float16* Eb = E   + (size_t)b * (128 * 512);
  const _Float16* Ab = A   + (size_t)b * (128 * 128);
  const _Float16* Tb = TT  + (size_t)b * (128 * 512);
  const _Float16* Ub = UT  + (size_t)b * (128 * 128);
  const _Float16* Xb = Xin + (size_t)b * (128 * 128);

  __builtin_prefetch(Eb + (size_t)i0 * 512, 0, 1);

  v8f acc[8] = {};
  gemm2_term(Eb, Tb,  512, 16, Bs, acc, tid);      // edge term,  K=512
  gemm2_term(Ab, Ub,  128, 4,  Bs, acc, tid);      // kbias term, K=128
  gemm2_term(Xb, RwT, 128, 4,  Bs, acc, tid);      // root term,  K=128

  const int l  = lane & 15;
  const int hi = lane >> 4;
  const int ib = i0 + hi * 8;
#pragma unroll
  for (int c = 0; c < 8; ++c) {
    const int o  = c * 16 + l;
    const float bv = bias[o];
#pragma unroll
    for (int v = 0; v < 8; ++v) {
      float val = acc[c][v] + bv;
      val = val > 0.f ? val : 0.f;
      Xout[((size_t)b * 128 + (ib + v)) * 128 + o] = (_Float16)val;
    }
  }
}

// ------------------------------------------------------------------
// Prep kernels: weight packing (f32 -> f16, transposed, K zero-padded),
// input casts.
// ------------------------------------------------------------------
__global__ void __launch_bounds__(256)
prep_weights(const float* __restrict__ kw, const float* __restrict__ kb,
             const float* __restrict__ rw, int f_in,
             _Float16* __restrict__ WT, _Float16* __restrict__ RwT) {
  int idx = blockIdx.x * blockDim.x + threadIdx.x;   // [0, 98304)
  if (idx < 640 * 128) {
    int c = idx >> 7, f = idx & 127;
    float v = 0.f;
    if (f < f_in) {
      if (c < 512) { int s = c >> 7, o = c & 127;
                     v = kw[(size_t)s * FH * f_in + (size_t)o * f_in + f]; }
      else         { int o = c - 512;
                     v = kb[(size_t)o * f_in + f]; }
    }
    WT[idx] = (_Float16)v;
  } else {
    int r = idx - 640 * 128, o = r >> 7, f = r & 127;
    float v = (f < f_in) ? rw[(size_t)f * FH + o] : 0.f;
    RwT[(size_t)o * FH + f] = (_Float16)v;
  }
}

__global__ void __launch_bounds__(256)
prep_x0(const float* __restrict__ x, _Float16* __restrict__ X0) {
  int idx = blockIdx.x * blockDim.x + threadIdx.x;   // 32768*128
  int m = idx >> 7, f = idx & 127;
  X0[idx] = (_Float16)((f < FNODE) ? x[(size_t)m * (FNODE + 1) + f] : 0.f);
}

__global__ void __launch_bounds__(256)
prep_e(const float* __restrict__ e, const float* __restrict__ a,
       _Float16* __restrict__ Ef) {
  size_t idx = (size_t)blockIdx.x * blockDim.x + threadIdx.x;  // 16.7M
  size_t bi = idx >> 9;
  int j = ((int)idx & 511) >> 2;
  Ef[idx] = (_Float16)(e[idx] * a[bi * 128 + j]);
}

__global__ void __launch_bounds__(256)
prep_a(const float* __restrict__ a, _Float16* __restrict__ Af) {
  size_t idx = (size_t)blockIdx.x * blockDim.x + threadIdx.x;  // 4.2M
  Af[idx] = (_Float16)a[idx];
}

// ------------------------------------------------------------------
// Masked pool + dense head:  out[b] = (sum_n h[b,n,:]*mask[b,n]) @ dw + db
// ------------------------------------------------------------------
__global__ void __launch_bounds__(128)
pool_dense(const _Float16* __restrict__ H, const float* __restrict__ x,
           const float* __restrict__ dw, const float* __restrict__ db,
           float* __restrict__ out) {
  __shared__ float pooled[FH];
  const int b = blockIdx.x, o = threadIdx.x;
  float s = 0.f;
  for (int n = 0; n < NN; ++n) {
    float m = x[((size_t)b * NN + n) * (FNODE + 1) + FNODE];
    s += (float)H[((size_t)b * NN + n) * FH + o] * m;
  }
  pooled[o] = s;
  __syncthreads();
  if (o < NOUT) {
    float acc = db[o];
    for (int f = 0; f < FH; ++f) acc += pooled[f] * dw[f * NOUT + o];
    out[b * NOUT + o] = acc;
  }
}

// ------------------------------------------------------------------
// Host launcher
// ------------------------------------------------------------------
extern "C" void kernel_launch(void* const* d_in, const int* in_sizes, int n_in,
                              void* d_out, int out_size, void* d_ws, size_t ws_size,
                              hipStream_t stream) {
  const float* x  = (const float*)d_in[0];
  const float* a  = (const float*)d_in[1];
  const float* e  = (const float*)d_in[2];
  const float* knw[4] = {(const float*)d_in[3],  (const float*)d_in[4],
                         (const float*)d_in[5],  (const float*)d_in[6]};
  const float* knb[4] = {(const float*)d_in[7],  (const float*)d_in[8],
                         (const float*)d_in[9],  (const float*)d_in[10]};
  const float* rtw[4] = {(const float*)d_in[11], (const float*)d_in[12],
                         (const float*)d_in[13], (const float*)d_in[14]};
  const float* bia[4] = {(const float*)d_in[15], (const float*)d_in[16],
                         (const float*)d_in[17], (const float*)d_in[18]};
  const float* dw = (const float*)d_in[19];
  const float* db = (const float*)d_in[20];
  float* out = (float*)d_out;

  char* ws = (char*)d_ws;
  const size_t SZ_X  = (size_t)32768 * 128 * 2;      //  8 MB
  const size_t SZ_TT = (size_t)NB * 128 * 512 * 2;   // 33 MB
  const size_t SZ_UT = (size_t)NB * 128 * 128 * 2;   //  8 MB
  _Float16* X0  = (_Float16*)(ws);
  _Float16* X1  = (_Float16*)(ws + SZ_X);
  _Float16* TT  = (_Float16*)(ws + 2 * SZ_X);
  _Float16* UT  = (_Float16*)(ws + 2 * SZ_X + SZ_TT);
  _Float16* Ef  = (_Float16*)(ws + 2 * SZ_X + SZ_TT + SZ_UT);
  _Float16* Af  = (_Float16*)(ws + 2 * SZ_X + 2 * SZ_TT + SZ_UT);
  _Float16* WT  = (_Float16*)(ws + 2 * SZ_X + 2 * SZ_TT + 2 * SZ_UT);
  _Float16* RwT = (_Float16*)(ws + 2 * SZ_X + 2 * SZ_TT + 2 * SZ_UT +
                              (size_t)4 * 640 * 128 * 2);

  const int f_ins[4] = {FNODE, FH, FH, FH};
  for (int l = 0; l < 4; ++l)
    prep_weights<<<384, 256, 0, stream>>>(knw[l], knb[l], rtw[l], f_ins[l],
                                          WT + (size_t)l * 640 * 128,
                                          RwT + (size_t)l * 128 * 128);
  prep_x0<<<16384, 256, 0, stream>>>(x, X0);
  prep_e<<<65536, 256, 0, stream>>>(e, a, Ef);
  prep_a<<<16384, 256, 0, stream>>>(a, Af);

  for (int l = 0; l < 4; ++l) {
    _Float16* Xin  = (l & 1) ? X1 : X0;
    _Float16* Xout = (l & 1) ? X0 : X1;
    gemm1_transform<<<2048, 256, 0, stream>>>(Xin, WT + (size_t)l * 640 * 128,
                                              TT, UT);
    gemm2_aggregate<<<NB, 256, 0, stream>>>(Ef, Af, TT, UT, Xin,
                                            RwT + (size_t)l * 128 * 128,
                                            bia[l], Xout);
  }
  // Chain X0->X1->X0->X1->X0: final hidden state is in X0.
  pool_dense<<<NB, 128, 0, stream>>>(X0, x, dw, db, out);
}
